// GumbelSampler_66039417143487
// MI455X (gfx1250) — compile-verified, matrix operand order checked by reading, not run on hardware
//
#include <hip/hip_runtime.h>

typedef __attribute__((ext_vector_type(16))) _Float16 v16h;
typedef __attribute__((ext_vector_type(8)))  float    v8f;

#define NMAX    128
#define NN      (NMAX * NMAX)        // 16384
#define BSZ     64
#define ENS     2
#define TENS    2
#define ROWS    (TENS * BSZ * ENS)   // 256
#define TOPK    64
#define TAU_INV 10.0f
#define EPS_TINY 1.17549435e-38f
#define NEG_BIG (-3.402823466e+38f)
#define THREADS 1024
#define VPT     (NN / THREADS)       // 16 elements per thread
#define WAVES   (THREADS / 32)       // 32 waves (wave32)

// ---------------------------------------------------------------------------
// Cross-lane XOR exchange. masks 1/2/4/8 stay inside a 16-lane group ->
// v_permlane16_b32 (VALU, no LDS round-trip, no dscnt stall). mask 16 ->
// ds_swizzle SWAPX16 (one DS op, no index VGPR). Guarded fallbacks keep the
// compile safe if a builtin is unavailable on this toolchain.
// ---------------------------------------------------------------------------
__device__ __forceinline__ unsigned xor_u32(unsigned x, int mask) {
#if __has_builtin(__builtin_amdgcn_permlane16) && __has_builtin(__builtin_amdgcn_ds_swizzle)
    switch (mask) {
    case 1:  return __builtin_amdgcn_permlane16(x, x, 0x67452301u, 0xEFCDAB89u, false, false);
    case 2:  return __builtin_amdgcn_permlane16(x, x, 0x54761032u, 0xDCFE98BAu, false, false);
    case 4:  return __builtin_amdgcn_permlane16(x, x, 0x32107654u, 0xBA98FEDCu, false, false);
    case 8:  return __builtin_amdgcn_permlane16(x, x, 0xFEDCBA98u, 0x76543210u, false, false);
    default: return (unsigned)__builtin_amdgcn_ds_swizzle((int)x, 0x401F); // SWAPX16
    }
#else
    return (unsigned)__shfl_xor((int)x, mask, 32);
#endif
}
__device__ __forceinline__ float xor_f32(float v, int mask) {
    return __uint_as_float(xor_u32(__float_as_uint(v), mask));
}

__device__ __forceinline__ float fast_rcp(float x) {
#if __has_builtin(__builtin_amdgcn_rcpf)
    return __builtin_amdgcn_rcpf(x);
#else
    return 1.0f / x;
#endif
}

// wave32 max all-reduce: 1 DS swizzle + 4 permlane16
__device__ __forceinline__ float wave_reduce_max(float v) {
    v = fmaxf(v, xor_f32(v, 16));
    v = fmaxf(v, xor_f32(v,  8));
    v = fmaxf(v, xor_f32(v,  4));
    v = fmaxf(v, xor_f32(v,  2));
    v = fmaxf(v, xor_f32(v,  1));
    return v;
}

// ---------------------------------------------------------------------------
// wave32 sum all-reduce through the matrix core:
//   A = ones(16x32), B holds each lane's partial as an f16 hi/lo pair.
//   D[m,n] = sum_k B[k,n]  => every D row is the column-sum vector; the two
//   16-lane halves of d[0] are identical, so only xor 1/2/4/8 (permlane16,
//   pure VALU) finish the all-reduce. f32 accuracy preserved by the two-term
//   f16 expansion; accumulation happens in the f32 C/D path.
// Must be called from uniform control flow (WMMA requires EXEC all-ones).
// ---------------------------------------------------------------------------
__device__ __forceinline__ float wmma_allreduce_sum(float x) {
    _Float16 hi = (_Float16)x;
    _Float16 lo = (_Float16)(x - (float)hi);
    v16h a, bm;
#pragma unroll
    for (int i = 0; i < 16; ++i) { a[i] = (_Float16)1.0f; bm[i] = (_Float16)0.0f; }
    bm[0] = hi;
    bm[1] = lo;
    v8f c = {};
    // (neg_a, A, neg_b, B, c_mod, C, reuse_a, reuse_b)
    v8f d = __builtin_amdgcn_wmma_f32_16x16x32_f16(false, a, false, bm,
                                                   (short)0, c, false, false);
    float s = d[0];
    s += xor_f32(s, 1);
    s += xor_f32(s, 2);
    s += xor_f32(s, 4);
    s += xor_f32(s, 8);
    return s; // every lane holds the 32-lane total
}

// ---------------------------------------------------------------------------
// One block per row r of flat (256 rows). Row fully resident in VGPRs.
// Element mapping: thread owns 8 adjacent column pairs:
//   c(k) = (tid<<1) + ((k>>1)<<11) + (k&1),  k in [0,16)
// so the 16 MB gumbel input is read with b64 vector loads.
// ---------------------------------------------------------------------------
__global__ __launch_bounds__(THREADS, 1)
void gumbel_topk_kernel(const float* __restrict__ scores,
                        const float* __restrict__ gumbel,
                        float* __restrict__ out) {
    __shared__ float s_max[WAVES];
    __shared__ float s_sum[WAVES];
    __shared__ float s_tv[2][WAVES];
    __shared__ int   s_ti[2][WAVES];

    const int tid  = threadIdx.x;
    const int lane = tid & 31;
    const int wave = tid >> 5;

    const int r = blockIdx.x;      // row in flat: r = t*128 + (b*2 + e)
    const int t = r >> 7;
    const int q = r & 127;
    const int b = q >> 1;
    const int e = q & 1;

    float f[VPT], oh[VPT], kh[VPT];

    // flat[r, c] = scores[b, i, j, e] + gumbel[r, c], c = i*N + j
    const size_t sbase = (size_t)b * (size_t)(NN * ENS) + (size_t)e;
    const size_t gbase = (size_t)r * (size_t)NN;
#pragma unroll
    for (int j = 0; j < VPT / 2; ++j) {
        const int c0 = (tid << 1) + (j << 11);
        const float2 g = *(const float2*)(gumbel + gbase + c0); // 8B aligned
        f[2 * j]     = scores[sbase + (size_t)c0 * ENS]       + g.x;
        f[2 * j + 1] = scores[sbase + (size_t)(c0 + 1) * ENS] + g.y;
        oh[2 * j] = 0.0f; oh[2 * j + 1] = 0.0f;
        kh[2 * j] = 0.0f; kh[2 * j + 1] = 0.0f;
    }

    // ---- 64 soft top-k iterations (serial dependence chain) ----
    for (int it = 0; it < TOPK; ++it) {
        // f += log(max(1 - onehot, tiny)); iteration 0 adds log(1) == 0
        if (it != 0) {
#pragma unroll
            for (int k = 0; k < VPT; ++k)
                f[k] += __logf(fmaxf(1.0f - oh[k], EPS_TINY));
        }

        // block-wide max of f (softmax stabilization)
        float m = NEG_BIG;
#pragma unroll
        for (int k = 0; k < VPT; ++k) m = fmaxf(m, f[k]);
        m = wave_reduce_max(m);
        if (lane == 0) s_max[wave] = m;
        __syncthreads();
        m = wave_reduce_max(s_max[lane]);   // WAVES == 32 == lanes

        // exp((f - m)/tau) and block-wide sum (WMMA reduction)
        float ls = 0.0f;
#pragma unroll
        for (int k = 0; k < VPT; ++k) {
            const float ex = __expf((f[k] - m) * TAU_INV);
            oh[k] = ex;
            ls += ex;
        }
        const float wsum = wmma_allreduce_sum(ls);      // per-wave total
        if (lane == 0) s_sum[wave] = wsum;
        __syncthreads();
        const float total = wmma_allreduce_sum(s_sum[lane]); // block total
        // (next write to s_max is fenced by this iteration's second barrier;
        //  next write to s_sum is fenced by next iteration's first barrier)

        const float inv = fast_rcp(total);
#pragma unroll
        for (int k = 0; k < VPT; ++k) {
            oh[k] *= inv;       // softmax
            kh[k] += oh[k];     // khot accumulate
        }
    }

    // ---- hard top-64 of khot: 64 rounds of block argmax (ping-pong LDS) ----
    float w[VPT];
#pragma unroll
    for (int k = 0; k < VPT; ++k) w[k] = kh[k];
    unsigned int hard = 0u;

    for (int it = 0; it < TOPK; ++it) {
        float bv = NEG_BIG;
        int   bi = 0x7FFFFFFF;
#pragma unroll
        for (int k = 0; k < VPT; ++k) {
            const float v   = w[k];
            const int   idx = (tid << 1) + ((k >> 1) << 11) + (k & 1); // column c
            const bool better = (v > bv) || (v == bv && idx < bi);
            bv = better ? v   : bv;
            bi = better ? idx : bi;
        }
#pragma unroll
        for (int off = 16; off >= 1; off >>= 1) {
            const float ov = xor_f32(bv, off);
            const int   oi = (int)xor_u32((unsigned)bi, off);
            const bool better = (ov > bv) || (ov == bv && oi < bi);
            bv = better ? ov : bv;
            bi = better ? oi : bi;
        }
        const int pb = it & 1;
        if (lane == 0) { s_tv[pb][wave] = bv; s_ti[pb][wave] = bi; }
        __syncthreads();
        bv = s_tv[pb][lane];
        bi = s_ti[pb][lane];
#pragma unroll
        for (int off = 16; off >= 1; off >>= 1) {
            const float ov = xor_f32(bv, off);
            const int   oi = (int)xor_u32((unsigned)bi, off);
            const bool better = (ov > bv) || (ov == bv && oi < bi);
            bv = better ? ov : bv;
            bi = better ? oi : bi;
        }
        // all threads agree on winning column bi; owner marks it
        const bool mine = (((bi >> 1) & (THREADS - 1)) == tid);
        const int  kk   = (((bi >> 11) << 1) | (bi & 1));
        if (mine) hard |= (1u << kk);
#pragma unroll
        for (int j2 = 0; j2 < VPT; ++j2)
            w[j2] = (mine && (j2 == kk)) ? NEG_BIG : w[j2];
        // ping-pong buffers: next round writes the other bank, so the next
        // barrier (top of next round) is the only fence needed
    }

    // ---- straight-through output, permuted to (t, b, i, j, e) ----
    const size_t obase = (size_t)(t * BSZ + b) * (size_t)(NN * ENS);
#pragma unroll
    for (int k = 0; k < VPT; ++k) {
        const int   c   = (tid << 1) + ((k >> 1) << 11) + (k & 1);
        const float khv = kh[k];
        const float hv  = ((hard >> k) & 1u) ? 1.0f : 0.0f;
        out[obase + (size_t)c * ENS + e] = (hv - khv) + khv; // matches eval order
    }
}

extern "C" void kernel_launch(void* const* d_in, const int* in_sizes, int n_in,
                              void* d_out, int out_size, void* d_ws, size_t ws_size,
                              hipStream_t stream) {
    (void)in_sizes; (void)n_in; (void)out_size; (void)d_ws; (void)ws_size;
    const float* scores = (const float*)d_in[0];
    const float* gumbel = (const float*)d_in[1];
    float* out = (float*)d_out;
    hipLaunchKernelGGL(gumbel_topk_kernel, dim3(ROWS), dim3(THREADS), 0, stream,
                       scores, gumbel, out);
}
